// L3UnfreezeModel_16982300688932
// MI455X (gfx1250) — matrix operand order repaired and checked by reading, not compiled, not run on hardware
//
#include <hip/hip_runtime.h>
#include <hip/hip_bf16.h>

// ---------------- problem constants ----------------
constexpr int Bc   = 64;
constexpr int Tc   = 196;
constexpr int Dc   = 2048;
constexpr int Hc   = 32;
constexpr int KVHc = 8;
constexpr int HDc  = 64;
constexpr int Ic   = 8192;
constexpr int QDc  = 512;
constexpr int PDc  = 128;
constexpr int TPAD = 224;            // T rounded up to 32 for safe vector loads
constexpr float THETA = 500000.0f;
constexpr float EPSc  = 1e-5f;

typedef __attribute__((ext_vector_type(16))) __bf16 v16bf;
typedef __attribute__((ext_vector_type(8)))  float  v8f;

union V16 { v16bf v; uint4 u[2]; };

constexpr int LDSS = 40;   // bf16 row stride in LDS (32 data + 8 pad)

// ================================================================
// Tensor Data Mover support (gfx1250). D# layout per CDNA5 ISA ch.8.
// This toolchain's builtin is the 6-arg form:
//   (u32x4 g0, i32x8 g1, i32x4 g2, i32x4 g3, i32x8 extra, i32 cpol)
// ================================================================
#if defined(__HIP_DEVICE_COMPILE__) && __has_builtin(__builtin_amdgcn_tensor_load_to_lds) && __has_builtin(__builtin_amdgcn_s_wait_tensorcnt)
#define HAVE_TDM 1
typedef unsigned int u32x4 __attribute__((ext_vector_type(4)));
typedef int          i32x4 __attribute__((ext_vector_type(4)));
typedef int          i32x8 __attribute__((ext_vector_type(8)));

// Low 32 bits of a generic LDS pointer are the LDS byte offset (ISA 10.2).
__device__ __forceinline__ unsigned lds_off_u32(const void* p) {
  return (unsigned)(unsigned long long)(size_t)p;
}

// DMA a (tile_dim1=128 rows) x (tile_dim0=32 elems) bf16 tile into LDS,
// padding each 64B row with 16B so LDS rows have stride LDSS(=40) bf16.
// tensor_dim0/1 = elements remaining from the tile start -> OOB rows/cols
// are zero-filled by the TDM (handles M tails with no guards).
__device__ __forceinline__ void tdm_load_tile(const __bf16* gbase,
                                              unsigned lds_off,
                                              int rows_rem, int k_rem,
                                              long long ldK) {
  unsigned long long ga = (unsigned long long)(size_t)gbase;
  u32x4 g0;
  g0[0] = 1u;                                        // count=1 (user desc)
  g0[1] = lds_off;                                   // lds_addr [63:32]
  g0[2] = (unsigned)(ga & 0xFFFFFFFFu);              // global_addr lo
  g0[3] = (unsigned)((ga >> 32) & 0x1FFFFFFu)        // global_addr hi (57b)
        | (2u << 30);                                // type=2 ("image")
  unsigned td0 = (unsigned)k_rem;                    // tensor_dim0
  unsigned td1 = (unsigned)rows_rem;                 // tensor_dim1
  unsigned long long s0 = (unsigned long long)ldK;   // tensor_dim0_stride
  i32x8 g1;
  g1[0] = (int)((1u << 16)        // data_size = 1 -> 2-byte elements
              | (1u << 20)        // pad_enable
              | (3u << 22)        // pad_interval: 16 DWORDs (one 64B row)
              | (3u << 25));      // pad_amount:  4 DWORDs (16B -> stride 80B)
  g1[1] = (int)((td0 & 0xFFFFu) << 16);              // [63:48] tdim0 lo16
  g1[2] = (int)((td0 >> 16) | ((td1 & 0xFFFFu) << 16));
  g1[3] = (int)((td1 >> 16) | (32u << 16));          // [127:112] tile_dim0=32
  g1[4] = (int)128;                                  // [143:128] tile_dim1=128
  g1[5] = (int)(unsigned)(s0 & 0xFFFFFFFFu);         // stride0 lo32
  g1[6] = (int)(unsigned)((s0 >> 32) & 0xFFFFu);     // stride0 hi16
  g1[7] = 0;
  i32x4 gz4 = {0, 0, 0, 0};                          // groups 2/3: dims unused
  i32x8 gz8 = {0, 0, 0, 0, 0, 0, 0, 0};
  __builtin_amdgcn_tensor_load_to_lds(g0, g1, gz4, gz4, gz8, 0);
}
#endif

// ================================================================
// Weight prep: fp32 W[K,N] -> bf16 Wt[N,K]  (LDS-tiled transpose)
// ================================================================
__global__ __launch_bounds__(256) void transpose_to_bf16(
    const float* __restrict__ W, __bf16* __restrict__ Wt, int K, int N) {
  __shared__ float tile[32][33];
  int n0 = blockIdx.x * 32, k0 = blockIdx.y * 32;
  int tx = threadIdx.x, ty = threadIdx.y;          // block (32, 8)
#pragma unroll
  for (int i = 0; i < 32; i += 8)
    tile[ty + i][tx] = W[(size_t)(k0 + ty + i) * N + n0 + tx];
  __syncthreads();
#pragma unroll
  for (int i = 0; i < 32; i += 8)
    Wt[(size_t)(n0 + ty + i) * K + k0 + tx] = (__bf16)tile[tx][ty + i];
}

// ================================================================
// RMSNorm: fp32 [R,D] * w[D] -> bf16 [R,D]
// ================================================================
__global__ __launch_bounds__(256) void rmsnorm_bf16_k(
    const float* __restrict__ x, const float* __restrict__ w,
    __bf16* __restrict__ out, int D) {
  int row = blockIdx.x, tid = threadIdx.x;
  const float* xr = x + (size_t)row * D;
  float ss = 0.f;
  for (int i = tid; i < D / 4; i += 256) {
    float4 v = ((const float4*)xr)[i];
    ss += v.x * v.x + v.y * v.y + v.z * v.z + v.w * v.w;
  }
#pragma unroll
  for (int d = 1; d < 32; d <<= 1) ss += __shfl_xor(ss, d, 32);
  __shared__ float warps[8];
  if ((tid & 31) == 0) warps[tid >> 5] = ss;
  __syncthreads();
  float tot = 0.f;
#pragma unroll
  for (int i = 0; i < 8; ++i) tot += warps[i];
  float rs = rsqrtf(tot / (float)D + EPSc);
  __bf16* orow = out + (size_t)row * D;
  for (int i = tid; i < D; i += 256) orow[i] = (__bf16)(xr[i] * rs * w[i]);
}

// ================================================================
// Generic bf16 WMMA GEMM: C[M,N] = A[M,K] @ Bt[N,K]^T (+resid)
// 256 thr = 8 waves; tile 128x128; wave 32x64; K-step 32;
// double-buffered LDS fed by the Tensor Data Mover (fallback: vector loads)
// ================================================================
__device__ __forceinline__ void gemm_compute_tile(
    const __bf16* __restrict__ Asb, const __bf16* __restrict__ Bsb,
    int wm, int wn, int n16, int half, v8f acc[2][4]) {
  V16 a[2], bb[4];
  const int abase = half * 8;                   // A: {b..b+8, b+16..b+24}
  const int kb = half * 8;                      // B: {b..b+8, b+16..b+24}
#pragma unroll
  for (int mi = 0; mi < 2; ++mi) {
    int r = wm * 32 + mi * 16 + n16;
    a[mi].u[0] = *(const uint4*)(&Asb[r * LDSS + abase]);
    a[mi].u[1] = *(const uint4*)(&Asb[r * LDSS + abase + 16]);
  }
#pragma unroll
  for (int ni = 0; ni < 4; ++ni) {
    int r = wn * 64 + ni * 16 + n16;
    bb[ni].u[0] = *(const uint4*)(&Bsb[r * LDSS + kb]);
    bb[ni].u[1] = *(const uint4*)(&Bsb[r * LDSS + kb + 16]);
  }
#pragma unroll
  for (int mi = 0; mi < 2; ++mi)
#pragma unroll
    for (int ni = 0; ni < 4; ++ni)
      acc[mi][ni] = __builtin_amdgcn_wmma_f32_16x16x32_bf16(
          false, a[mi].v, false, bb[ni].v, (short)0, acc[mi][ni], false, false);
}

__global__ __launch_bounds__(256) void gemm_bf16(
    const __bf16* __restrict__ A, const __bf16* __restrict__ Bt,
    const float* __restrict__ resid, float* __restrict__ outF,
    __bf16* __restrict__ outB, int M, int N, int K) {
  __shared__ alignas(16) __bf16 AsB[2][128 * LDSS];
  __shared__ alignas(16) __bf16 BsB[2][128 * LDSS];

  const int tid = threadIdx.x, lane = tid & 31, w = tid >> 5;
  const int wm = w & 3, wn = w >> 2;
  const int bm0 = blockIdx.y * 128, bn0 = blockIdx.x * 128;
  const int half = lane >> 4, n16 = lane & 15;

  const v8f vzero = {0.f, 0.f, 0.f, 0.f, 0.f, 0.f, 0.f, 0.f};
  v8f acc[2][4];
#pragma unroll
  for (int mi = 0; mi < 2; ++mi)
#pragma unroll
    for (int ni = 0; ni < 4; ++ni) acc[mi][ni] = vzero;

#if defined(HAVE_TDM)
  // ---- TDM-fed, double-buffered pipeline ----
  if (w == 0) {
    tdm_load_tile(A + (size_t)bm0 * K, lds_off_u32(&AsB[0][0]),
                  M - bm0, K, K);
    tdm_load_tile(Bt + (size_t)bn0 * K, lds_off_u32(&BsB[0][0]),
                  N - bn0, K, K);
    __builtin_amdgcn_s_wait_tensorcnt(0);
  }
  __syncthreads();
  int cur = 0;
  for (int k0 = 0; k0 < K; k0 += 32) {
    const bool have_next = (k0 + 32 < K);
    if (w == 0 && have_next) {                  // DMA next tiles, overlapped
      tdm_load_tile(A + (size_t)bm0 * K + (k0 + 32),
                    lds_off_u32(&AsB[cur ^ 1][0]), M - bm0, K - (k0 + 32), K);
      tdm_load_tile(Bt + (size_t)bn0 * K + (k0 + 32),
                    lds_off_u32(&BsB[cur ^ 1][0]), N - bn0, K - (k0 + 32), K);
    }
    gemm_compute_tile(&AsB[cur][0], &BsB[cur][0], wm, wn, n16, half, acc);
    if (w == 0 && have_next) __builtin_amdgcn_s_wait_tensorcnt(0);
    __syncthreads();
    cur ^= 1;
  }
#else
  // ---- fallback: double-buffered vector loads ----
  uint4 da[2], db[2];
  auto load_tile = [&](int k0) {
#pragma unroll
    for (int it = 0; it < 2; ++it) {
      int i = tid + it * 256;
      int r = i >> 2, ch = (i & 3) * 8;
      int gm = bm0 + r;
      da[it] = make_uint4(0u, 0u, 0u, 0u);
      if (gm < M) {
        da[it] = *(const uint4*)(A + (size_t)gm * K + k0 + ch);
        __builtin_prefetch(A + (size_t)gm * K + k0 + 64 + ch, 0, 1);
      }
      int gn = bn0 + r;
      db[it] = make_uint4(0u, 0u, 0u, 0u);
      if (gn < N) {
        db[it] = *(const uint4*)(Bt + (size_t)gn * K + k0 + ch);
        __builtin_prefetch(Bt + (size_t)gn * K + k0 + 64 + ch, 0, 1);
      }
    }
  };
  auto store_tile = [&](int buf) {
#pragma unroll
    for (int it = 0; it < 2; ++it) {
      int i = tid + it * 256;
      int r = i >> 2, ch = (i & 3) * 8;
      *(uint4*)(&AsB[buf][r * LDSS + ch]) = da[it];
      *(uint4*)(&BsB[buf][r * LDSS + ch]) = db[it];
    }
  };
  load_tile(0);
  store_tile(0);
  __syncthreads();
  int cur = 0;
  for (int k0 = 0; k0 < K; k0 += 32) {
    const bool have_next = (k0 + 32 < K);
    if (have_next) load_tile(k0 + 32);          // global loads in flight
    gemm_compute_tile(&AsB[cur][0], &BsB[cur][0], wm, wn, n16, half, acc);
    if (have_next) store_tile(cur ^ 1);
    __syncthreads();
    cur ^= 1;
  }
#endif

  // ---- epilogue ----
#pragma unroll
  for (int mi = 0; mi < 2; ++mi)
#pragma unroll
    for (int ni = 0; ni < 4; ++ni)
#pragma unroll
      for (int r = 0; r < 8; ++r) {
        int row = bm0 + wm * 32 + mi * 16 + half * 8 + r;
        int col = bn0 + wn * 64 + ni * 16 + n16;
        if (row < M && col < N) {
          float v = acc[mi][ni][r];
          if (resid) v += resid[(size_t)row * N + col];
          if (outF) outF[(size_t)row * N + col] = v;
          if (outB) outB[(size_t)row * N + col] = (__bf16)v;
        }
      }
}

// ================================================================
// RoPE: fp32 [B*T, nh*HD] -> bf16 [B, nh, T, HD]
// ================================================================
__global__ __launch_bounds__(32) void rope_kernel(
    const float* __restrict__ src, __bf16* __restrict__ dst, int nh) {
  int d = threadIdx.x;            // 0..31 (pairs d, d+32)
  int hh = blockIdx.x, t = blockIdx.y, b = blockIdx.z;
  const float* s = src + ((size_t)(b * Tc + t) * nh + hh) * HDc;
  float x1 = s[d], x2 = s[d + 32];
  float inv = __expf(-((float)(2 * d) / (float)HDc) * __logf(THETA));
  float fr = (float)t * inv;
  float sn, cs;
  __sincosf(fr, &sn, &cs);
  __bf16* o = dst + (((size_t)(b * nh + hh)) * Tc + t) * HDc;
  o[d]      = (__bf16)(x1 * cs - x2 * sn);
  o[d + 32] = (__bf16)(x2 * cs + x1 * sn);
}

// V: fp32 [B*T, KVH*HD] -> bf16 [B, KVH, HD, TPAD] (transposed for PV WMMA)
__global__ __launch_bounds__(256) void v_layout_k(
    const float* __restrict__ v, __bf16* __restrict__ vT) {
  size_t idx = (size_t)blockIdx.x * 256 + threadIdx.x;
  size_t total = (size_t)Bc * KVHc * HDc * Tc;
  if (idx >= total) return;
  int t = idx % Tc;
  int d = (idx / Tc) % HDc;
  int kvh = (idx / ((size_t)Tc * HDc)) % KVHc;
  int b = idx / ((size_t)Tc * HDc * KVHc);
  float val = v[((size_t)(b * Tc + t)) * (KVHc * HDc) + kvh * HDc + d];
  vT[(((size_t)(b * KVHc + kvh)) * HDc + d) * TPAD + t] = (__bf16)val;
}

// ================================================================
// Flash attention: 1 wave per (b, h, 16-row q tile); kv steps of 32
// ================================================================
__global__ __launch_bounds__(32) void flash_attn_k(
    const __bf16* __restrict__ qb, const __bf16* __restrict__ kb_,
    const __bf16* __restrict__ vT, float* __restrict__ o_out) {
  __shared__ alignas(16) __bf16 pl[16 * LDSS];

  const int lane = threadIdx.x, half = lane >> 4, n16 = lane & 15;
  const int qt = blockIdx.x, h = blockIdx.y, b = blockIdx.z;
  const int kvh = h >> 2;                               // GQA: rep = H/KVH = 4

  const __bf16* Qb = qb + ((size_t)(b * Hc + h)) * Tc * HDc;
  const __bf16* Kb = kb_ + ((size_t)(b * KVHc + kvh)) * Tc * HDc;
  const __bf16* Vb = vT + ((size_t)(b * KVHc + kvh)) * HDc * TPAD;

  const int abase = half * 8, kb16 = half * 16;
  const int qrow = min(qt * 16 + n16, Tc - 1);

  V16 aQ[2];
#pragma unroll
  for (int dc = 0; dc < 2; ++dc) {
    const __bf16* p = Qb + (size_t)qrow * HDc + dc * 32;
    aQ[dc].u[0] = *(const uint4*)(p + abase);
    aQ[dc].u[1] = *(const uint4*)(p + abase + 16);
  }

  const v8f vzero = {0.f, 0.f, 0.f, 0.f, 0.f, 0.f, 0.f, 0.f};
  v8f oacc[4];
#pragma unroll
  for (int t = 0; t < 4; ++t) oacc[t] = vzero;
  float m[8], l[8];
#pragma unroll
  for (int r = 0; r < 8; ++r) { m[r] = -1e30f; l[r] = 0.f; }

  const int qmax = qt * 16 + 15;
  const int jend = min(qmax / 32 + 1, (Tc + 31) / 32);

  for (int j = 0; j < jend; ++j) {
    // ---- S = Q @ K^T (two 16-col tiles) ----
    v8f s0 = vzero, s1 = vzero;
    int kc0 = min(j * 32 + n16, Tc - 1);
    int kc1 = min(j * 32 + 16 + n16, Tc - 1);
#pragma unroll
    for (int dc = 0; dc < 2; ++dc) {
      V16 b0, b1;
      const __bf16* p0 = Kb + (size_t)kc0 * HDc + dc * 32 + abase;
      const __bf16* p1 = Kb + (size_t)kc1 * HDc + dc * 32 + abase;
      b0.u[0] = *(const uint4*)p0; b0.u[1] = *(const uint4*)(p0 + 16);
      b1.u[0] = *(const uint4*)p1; b1.u[1] = *(const uint4*)(p1 + 16);
      s0 = __builtin_amdgcn_wmma_f32_16x16x32_bf16(false, aQ[dc].v, false, b0.v,
                                                   (short)0, s0, false, false);
      s1 = __builtin_amdgcn_wmma_f32_16x16x32_bf16(false, aQ[dc].v, false, b1.v,
                                                   (short)0, s1, false, false);
    }

    // ---- online softmax, emit P (bf16) into LDS ----
    float alpha[8];
#pragma unroll
    for (int r = 0; r < 8; ++r) {
      int qr = qt * 16 + half * 8 + r;
      int c0 = j * 32 + n16, c1 = c0 + 16;
      float v0 = s0[r] * 0.125f, v1 = s1[r] * 0.125f;
      if (qr >= Tc || c0 > qr || c0 >= Tc) v0 = -1e30f;
      if (qr >= Tc || c1 > qr || c1 >= Tc) v1 = -1e30f;
      float rmax = fmaxf(v0, v1);
#pragma unroll
      for (int d = 1; d < 16; d <<= 1) rmax = fmaxf(rmax, __shfl_xor(rmax, d, 32));
      float mn = fmaxf(m[r], rmax);
      float a = __expf(m[r] - mn);
      float p0 = (v0 <= -1e29f) ? 0.f : __expf(v0 - mn);
      float p1 = (v1 <= -1e29f) ? 0.f : __expf(v1 - mn);
      float rsum = p0 + p1;
#pragma unroll
      for (int d = 1; d < 16; d <<= 1) rsum += __shfl_xor(rsum, d, 32);
      l[r] = l[r] * a + rsum;
      m[r] = mn;
      alpha[r] = a;
      pl[(half * 8 + r) * LDSS + n16]      = (__bf16)p0;
      pl[(half * 8 + r) * LDSS + n16 + 16] = (__bf16)p1;
    }
#pragma unroll
    for (int t = 0; t < 4; ++t)
#pragma unroll
      for (int r = 0; r < 8; ++r) oacc[t][r] *= alpha[r];
    __syncthreads();

    // ---- P (A layout) @ V (B layout) ----
    V16 pA;
    pA.u[0] = *(const uint4*)(&pl[n16 * LDSS + abase]);
    pA.u[1] = *(const uint4*)(&pl[n16 * LDSS + abase + 16]);
#pragma unroll
    for (int t = 0; t < 4; ++t) {
      const __bf16* vp = Vb + (size_t)(t * 16 + n16) * TPAD + j * 32 + abase;
      V16 bV;
      bV.u[0] = *(const uint4*)vp; bV.u[1] = *(const uint4*)(vp + 16);
      oacc[t] = __builtin_amdgcn_wmma_f32_16x16x32_bf16(
          false, pA.v, false, bV.v, (short)0, oacc[t], false, false);
    }
    __syncthreads();
  }

  // ---- write O = acc / l into [B, T, H*HD] fp32 ----
#pragma unroll
  for (int r = 0; r < 8; ++r) {
    int qr = qt * 16 + half * 8 + r;
    if (qr < Tc) {
      float inv = 1.0f / l[r];
#pragma unroll
      for (int t = 0; t < 4; ++t)
        o_out[((size_t)(b * Tc + qr)) * (Hc * HDc) + h * HDc + t * 16 + n16] =
            oacc[t][r] * inv;
    }
  }
}

// ================================================================
// Elementwise helpers
// ================================================================
__global__ __launch_bounds__(256) void convert_f32_bf16(
    const float* __restrict__ in, __bf16* __restrict__ out, size_t n) {
  size_t i = (size_t)blockIdx.x * 256 + threadIdx.x;
  if (i < n) out[i] = (__bf16)in[i];
}

__global__ __launch_bounds__(256) void silu_mul_k(
    __bf16* __restrict__ g, const __bf16* __restrict__ u, size_t n) {
  size_t i = (size_t)blockIdx.x * 256 + threadIdx.x;
  if (i < n) {
    float gv = (float)g[i], uv = (float)u[i];
    g[i] = (__bf16)(gv / (1.0f + __expf(-gv)) * uv);
  }
}

// cosine head: out[b,t] = scale * <zp_hat, zq_hat> + bias
__global__ __launch_bounds__(128) void head_k(
    const float* __restrict__ zp, const float* __restrict__ zq,
    const float* __restrict__ scale, const float* __restrict__ bias,
    float* __restrict__ out) {
  int row = blockIdx.x, b = row / Tc, p = threadIdx.x;   // PDIM=128 threads
  float zpv = zp[(size_t)row * PDc + p];
  float zqv = zq[(size_t)b * PDc + p];
  float a = zpv * zpv, c = zqv * zqv, d = zpv * zqv;
#pragma unroll
  for (int mset = 1; mset < 32; mset <<= 1) {
    a += __shfl_xor(a, mset, 32);
    c += __shfl_xor(c, mset, 32);
    d += __shfl_xor(d, mset, 32);
  }
  __shared__ float sa[4], sc[4], sd[4];
  if ((p & 31) == 0) { sa[p >> 5] = a; sc[p >> 5] = c; sd[p >> 5] = d; }
  __syncthreads();
  if (p == 0) {
    float A = sa[0] + sa[1] + sa[2] + sa[3];
    float C = sc[0] + sc[1] + sc[2] + sc[3];
    float D = sd[0] + sd[1] + sd[2] + sd[3];
    out[row] = (*scale) * (D / (sqrtf(A) * sqrtf(C))) + (*bias);
  }
}

// ================================================================
// Host orchestration
// ================================================================
extern "C" void kernel_launch(void* const* d_in, const int* in_sizes, int n_in,
                              void* d_out, int out_size, void* d_ws, size_t ws_size,
                              hipStream_t stream) {
  (void)in_sizes; (void)n_in; (void)out_size; (void)ws_size;
  const size_t R = (size_t)Bc * Tc;

  // ---- workspace carve-out ----
  size_t off = 0;
  auto take = [&](size_t bytes) -> void* {
    void* p = (char*)d_ws + off;
    off = (off + bytes + 255) & ~(size_t)255;
    return p;
  };
  void* Wq_t  = take((size_t)Dc * (Hc * HDc) * 2);
  void* Wk_t  = take((size_t)Dc * (KVHc * HDc) * 2);
  void* Wv_t  = take((size_t)Dc * (KVHc * HDc) * 2);
  void* Wo_t  = take((size_t)(Hc * HDc) * Dc * 2);
  void* Wg_t  = take((size_t)Dc * Ic * 2);
  void* Wu_t  = take((size_t)Dc * Ic * 2);
  void* Wd_t  = take((size_t)Ic * Dc * 2);
  void* Wp_t  = take((size_t)Dc * PDc * 2);
  void* Wqh_t = take((size_t)QDc * PDc * 2);
  void* h_bf  = take(R * Dc * 2);            // reused: h_in / o_bf / h_post / xn
  void* q_f32 = take(R * (Hc * HDc) * 4);    // reused as o_f32 after RoPE
  void* k_f32 = take(R * (KVHc * HDc) * 4);
  void* v_f32 = take(R * (KVHc * HDc) * 4);
  void* q_bf  = take(R * (Hc * HDc) * 2);
  void* k_bf  = take(R * (KVHc * HDc) * 2);
  void* vT_bf = take((size_t)Bc * KVHc * HDc * TPAD * 2);
  void* x1    = take(R * Dc * 4);
  void* g_bf  = take(R * Ic * 2);            // reused in-place for silu(g)*u
  void* u_bf  = take(R * Ic * 2);
  void* x2    = take(R * Dc * 4);
  void* zp    = take(R * PDc * 4);
  void* q2_bf = take((size_t)Bc * QDc * 2);
  void* zq    = take((size_t)Bc * PDc * 4);

  auto launchT = [&](const void* W, void* Wt, int K, int N) {
    dim3 g(N / 32, K / 32), blk(32, 8);
    transpose_to_bf16<<<g, blk, 0, stream>>>((const float*)W, (__bf16*)Wt, K, N);
  };
  auto gemm = [&](const void* A, const void* Bt, const void* resid, void* outF,
                  void* outB, int M, int N, int K) {
    dim3 g(N / 128, (M + 127) / 128);
    gemm_bf16<<<g, 256, 0, stream>>>((const __bf16*)A, (const __bf16*)Bt,
                                     (const float*)resid, (float*)outF,
                                     (__bf16*)outB, M, N, K);
  };

  // ---- weight prep (bf16, pre-transposed) ----
  launchT(d_in[2],  Wq_t,  Dc, Hc * HDc);
  launchT(d_in[3],  Wk_t,  Dc, KVHc * HDc);
  launchT(d_in[4],  Wv_t,  Dc, KVHc * HDc);
  launchT(d_in[5],  Wo_t,  Hc * HDc, Dc);
  launchT(d_in[6],  Wg_t,  Dc, Ic);
  launchT(d_in[7],  Wu_t,  Dc, Ic);
  launchT(d_in[8],  Wd_t,  Ic, Dc);
  launchT(d_in[12], Wp_t,  Dc, PDc);
  launchT(d_in[13], Wqh_t, QDc, PDc);

  // ---- attention sublayer ----
  rmsnorm_bf16_k<<<(int)R, 256, 0, stream>>>((const float*)d_in[0],
                                             (const float*)d_in[9],
                                             (__bf16*)h_bf, Dc);
  gemm(h_bf, Wq_t, nullptr, q_f32, nullptr, (int)R, Hc * HDc, Dc);
  gemm(h_bf, Wk_t, nullptr, k_f32, nullptr, (int)R, KVHc * HDc, Dc);
  gemm(h_bf, Wv_t, nullptr, v_f32, nullptr, (int)R, KVHc * HDc, Dc);

  rope_kernel<<<dim3(Hc, Tc, Bc), 32, 0, stream>>>((const float*)q_f32,
                                                   (__bf16*)q_bf, Hc);
  rope_kernel<<<dim3(KVHc, Tc, Bc), 32, 0, stream>>>((const float*)k_f32,
                                                     (__bf16*)k_bf, KVHc);
  {
    size_t nv = (size_t)Bc * KVHc * HDc * Tc;
    v_layout_k<<<(int)((nv + 255) / 256), 256, 0, stream>>>((const float*)v_f32,
                                                            (__bf16*)vT_bf);
  }

  void* o_f32 = q_f32;   // q_f32 fully consumed by RoPE -> reuse for O
  flash_attn_k<<<dim3((Tc + 15) / 16, Hc, Bc), 32, 0, stream>>>(
      (const __bf16*)q_bf, (const __bf16*)k_bf, (const __bf16*)vT_bf,
      (float*)o_f32);

  {
    size_t no = R * (size_t)(Hc * HDc);
    convert_f32_bf16<<<(int)((no + 255) / 256), 256, 0, stream>>>(
        (const float*)o_f32, (__bf16*)h_bf, no);
  }
  gemm(h_bf, Wo_t, d_in[0], x1, nullptr, (int)R, Dc, Hc * HDc);  // x1 = x + O@Wo

  // ---- MLP sublayer (SwiGLU) ----
  rmsnorm_bf16_k<<<(int)R, 256, 0, stream>>>((const float*)x1,
                                             (const float*)d_in[10],
                                             (__bf16*)h_bf, Dc);
  gemm(h_bf, Wg_t, nullptr, nullptr, g_bf, (int)R, Ic, Dc);
  gemm(h_bf, Wu_t, nullptr, nullptr, u_bf, (int)R, Ic, Dc);
  {
    size_t nm = R * (size_t)Ic;
    silu_mul_k<<<(int)((nm + 255) / 256), 256, 0, stream>>>((__bf16*)g_bf,
                                                            (const __bf16*)u_bf, nm);
  }
  gemm(g_bf, Wd_t, x1, x2, nullptr, (int)R, Dc, Ic);             // x2 = x1 + mlp

  // ---- final norm + bilinear cosine head ----
  rmsnorm_bf16_k<<<(int)R, 256, 0, stream>>>((const float*)x2,
                                             (const float*)d_in[11],
                                             (__bf16*)h_bf, Dc);
  gemm(h_bf, Wp_t, nullptr, zp, nullptr, (int)R, PDc, Dc);

  {
    size_t nq = (size_t)Bc * QDc;
    convert_f32_bf16<<<(int)((nq + 255) / 256), 256, 0, stream>>>(
        (const float*)d_in[1], (__bf16*)q2_bf, nq);
  }
  gemm(q2_bf, Wqh_t, nullptr, zq, nullptr, Bc, PDc, QDc);

  head_k<<<(int)R, 128, 0, stream>>>((const float*)zp, (const float*)zq,
                                     (const float*)d_in[14],
                                     (const float*)d_in[15], (float*)d_out);
}
